// TransformerEncoderLayer_82815559402137
// MI455X (gfx1250) — matrix operand order repaired and checked by reading, not compiled, not run on hardware
//
#include <hip/hip_runtime.h>

// ---------------------------------------------------------------------------
// Transformer encoder layer (pre-LN, prior-weighted masked MHSA + FFN)
// for MI455X / gfx1250: all GEMMs via v_wmma_f32_16x16x32_f16 (wave32),
// ping-pong LDS staging via GLOBAL_LOAD_ASYNC_TO_LDS_B128 when available.
// B=4, L=2048, D=1024, H=16, DK=DV=64, DFF=4096.
// ---------------------------------------------------------------------------

typedef _Float16 half_t;
typedef __attribute__((ext_vector_type(8)))  _Float16 v8h;
typedef __attribute__((ext_vector_type(16))) _Float16 v16h;
typedef __attribute__((ext_vector_type(8)))  float    v8f;
typedef __attribute__((ext_vector_type(4)))  float    v4f;
typedef __attribute__((ext_vector_type(4)))  int      v4i;

#define Bc   4
#define Lc   2048
#define Dc   1024
#define Hc   16
#define DKc  64
#define DFFc 4096

enum { MODE_F16 = 0, MODE_SCORES = 1, MODE_F16_BIAS_RELU = 2, MODE_F32_RESID = 3 };

// block tile
#define TM 128
#define TN 64
#define TK 32
#define LDS_STRIDE 40   // 32 halves + 16B pad

#if __has_builtin(__builtin_amdgcn_global_load_async_to_lds_b128)
#define HAVE_ASYNC 1
#endif
#if __has_builtin(__builtin_amdgcn_s_wait_asynccnt)
#define HAVE_ASYNC_WAIT 1
#endif

#define AS1 __attribute__((address_space(1)))
#define AS3 __attribute__((address_space(3)))
typedef AS1 v4i* g4p;  // global int4*  (builtin's src type)
typedef AS3 v4i* l4p;  // LDS int4*     (builtin's dst type)

static __device__ __forceinline__ void wait_async_all() {
#if defined(HAVE_ASYNC_WAIT)
  __builtin_amdgcn_s_wait_asynccnt(0);
#elif defined(HAVE_ASYNC)
  asm volatile("s_wait_asynccnt 0x0" ::: "memory");
#endif
}

static __device__ __forceinline__ v16h cat16(v8h lo, v8h hi) {
  return __builtin_shufflevector(lo, hi, 0, 1, 2, 3, 4, 5, 6, 7,
                                         8, 9, 10, 11, 12, 13, 14, 15);
}

// ---------------------------------------------------------------------------
// Generic tiled WMMA GEMM:  C[m,n] = sum_k A[m,k] * W[n,k]   (A row-major,
// W in torch Linear layout [N,K]).  256 threads = 8 waves (4 in M x 2 in N),
// each wave computes 2x2 16x16 WMMA tiles -> 128x64 block tile, K-step 32,
// double-buffered LDS with async global->LDS copies overlapping the WMMAs.
// blockIdx.z decomposed as (zb = z/Hh, zh = z%Hh) with per-pointer strides.
// ---------------------------------------------------------------------------
template <int MODE, bool A32>
__global__ __launch_bounds__(256) void gemm_wmma(
    const void* __restrict__ Araw, long lda, long aSB, long aSH,
    const half_t* __restrict__ Wp, long ldw, long wSB, long wSH,
    float* __restrict__ Cf, long ldc, long cSB, long cSH,
    half_t* __restrict__ Ch, long chRow, long chCol, long chSB, long chSH,
    const float* __restrict__ bias,
    const float* __restrict__ resid, long ldr,
    const float* __restrict__ prior, const int* __restrict__ mask,
    int Hh, float scale, int M, int N, int K) {
  const int z  = blockIdx.z;
  const int zb = z / Hh;
  const int zh = z % Hh;

  const float*  Af  = nullptr;
  const half_t* A16 = nullptr;
  if constexpr (A32) Af  = (const float*)Araw  + (long)zb * aSB + (long)zh * aSH;
  else               A16 = (const half_t*)Araw + (long)zb * aSB + (long)zh * aSH;
  const half_t* W = Wp + (long)zb * wSB + (long)zh * wSH;

  const int m0 = blockIdx.y * TM;
  const int n0 = blockIdx.x * TN;

  __shared__ half_t As[2][TM * LDS_STRIDE];
  __shared__ half_t Ws[2][TN * LDS_STRIDE];

  const int tid    = threadIdx.x;
  const int lane   = tid & 31;
  const int wave   = tid >> 5;
  const int wm     = wave & 3;   // 4 waves along M (32 rows each)
  const int wn     = wave >> 2;  // 2 waves along N (32 cols each)
  const int halfId = lane >> 4;  // 0: lanes 0-15, 1: lanes 16-31
  const int l15    = lane & 15;

  // per-thread staging geometry: row = tid/4 (and +64 for 2nd A chunk),
  // 8-half (16B) chunk at column offset (tid%4)*8 -- K-loop invariant bases.
  const int srow = tid >> 2;
  const int sko  = (tid & 3) * 8;
  const float*  aSrcF0 = nullptr;
  const float*  aSrcF1 = nullptr;
  const half_t* aSrcH0 = nullptr;
  const half_t* aSrcH1 = nullptr;
  if constexpr (A32) {
    aSrcF0 = Af + (long)(m0 + srow) * lda + sko;
    aSrcF1 = Af + (long)(m0 + srow + 64) * lda + sko;
  } else {
    aSrcH0 = A16 + (long)(m0 + srow) * lda + sko;
    aSrcH1 = A16 + (long)(m0 + srow + 64) * lda + sko;
  }
  const half_t* wSrc = W + (long)(n0 + srow) * ldw + sko;
  half_t* aDst0[2] = { &As[0][srow * LDS_STRIDE + sko],        &As[1][srow * LDS_STRIDE + sko] };
  half_t* aDst1[2] = { &As[0][(srow + 64) * LDS_STRIDE + sko], &As[1][(srow + 64) * LDS_STRIDE + sko] };
  half_t* wDst[2]  = { &Ws[0][srow * LDS_STRIDE + sko],        &Ws[1][srow * LDS_STRIDE + sko] };

  auto stage = [&](int k0, int bb) {
    if constexpr (A32) {
      // attn fp32 -> f16 convert through VGPRs
      const v4f* s0 = (const v4f*)(aSrcF0 + k0);
      const v4f* s1 = (const v4f*)(aSrcF1 + k0);
      v4f f0 = s0[0], f1 = s0[1], f2 = s1[0], f3 = s1[1];
      v8h h0, h1;
      #pragma unroll
      for (int e = 0; e < 4; ++e) {
        h0[e] = (half_t)f0[e]; h0[4 + e] = (half_t)f1[e];
        h1[e] = (half_t)f2[e]; h1[4 + e] = (half_t)f3[e];
      }
      *(v8h*)aDst0[bb] = h0;
      *(v8h*)aDst1[bb] = h1;
#if defined(HAVE_ASYNC)
      __builtin_amdgcn_global_load_async_to_lds_b128(
          (g4p)(wSrc + k0), (l4p)wDst[bb], 0, 0);
#else
      *(v8h*)wDst[bb] = *(const v8h*)(wSrc + k0);
#endif
    } else {
#if defined(HAVE_ASYNC)
      __builtin_amdgcn_global_load_async_to_lds_b128(
          (g4p)(aSrcH0 + k0), (l4p)aDst0[bb], 0, 0);
      __builtin_amdgcn_global_load_async_to_lds_b128(
          (g4p)(aSrcH1 + k0), (l4p)aDst1[bb], 0, 0);
      __builtin_amdgcn_global_load_async_to_lds_b128(
          (g4p)(wSrc + k0), (l4p)wDst[bb], 0, 0);
#else
      *(v8h*)aDst0[bb] = *(const v8h*)(aSrcH0 + k0);
      *(v8h*)aDst1[bb] = *(const v8h*)(aSrcH1 + k0);
      *(v8h*)wDst[bb]  = *(const v8h*)(wSrc + k0);
      if (k0 + TK < K) {
        __builtin_prefetch(aSrcH0 + k0 + TK, 0, 3);
        __builtin_prefetch(wSrc + k0 + TK, 0, 3);
      }
#endif
    }
  };

  v8f acc[2][2] = {};

  // prologue: stage tile 0 into buffer 0
  stage(0, 0);
  wait_async_all();
  __syncthreads();

  int buf = 0;
  for (int k0 = 0; k0 < K; k0 += TK) {
    const int  nb   = buf ^ 1;
    const bool more = (k0 + TK) < K;
    if (more) stage(k0 + TK, nb);  // overlaps with WMMAs below

    // ---- fragments from current buffer ------------------------------------
    // A (16x32 f16): lane%16 = M row; lanes16-31 take K+8; VGPR0-3: K 0-15,
    // VGPR4-7: K 16-31  => two contiguous 16B LDS reads at kbase, kbase+16.
    v16h a[2];
    #pragma unroll
    for (int i = 0; i < 2; ++i) {
      const int row = wm * 32 + i * 16 + l15;
      v8h lo = *(const v8h*)&As[buf][row * LDS_STRIDE + halfId * 8];
      v8h hi = *(const v8h*)&As[buf][row * LDS_STRIDE + 16 + halfId * 8];
      a[i] = cat16(lo, hi);
    }
    // B (32x16 f16): lane%16 = N col; lanes16-31 take K+8; same K order as A
    // => two 16B LDS reads at kbase, kbase+16.
    v16h b[2];
    #pragma unroll
    for (int j = 0; j < 2; ++j) {
      const int col = wn * 32 + j * 16 + l15;
      v8h lo = *(const v8h*)&Ws[buf][col * LDS_STRIDE + halfId * 8];
      v8h hi = *(const v8h*)&Ws[buf][col * LDS_STRIDE + 16 + halfId * 8];
      b[j] = cat16(lo, hi);
    }

    #pragma unroll
    for (int i = 0; i < 2; ++i)
      #pragma unroll
      for (int j = 0; j < 2; ++j)
        acc[i][j] = __builtin_amdgcn_wmma_f32_16x16x32_f16(
            false, a[i], false, b[j], (short)0, acc[i][j], false, false);

    if (more) wait_async_all();  // next tile landed in LDS
    __syncthreads();             // publish next buffer / retire current
    buf = nb;
  }

  // ---- epilogue -----------------------------------------------------------
  const long cfBase = (long)zb * cSB + (long)zh * cSH;
  const long chBase = (long)zb * chSB + (long)zh * chSH;

  #pragma unroll
  for (int i = 0; i < 2; ++i) {
    #pragma unroll
    for (int j = 0; j < 2; ++j) {
      #pragma unroll
      for (int r = 0; r < 8; ++r) {
        const int m = m0 + wm * 32 + i * 16 + halfId * 8 + r;  // lanes16-31 -> M+8
        const int n = n0 + wn * 32 + j * 16 + l15;
        float v = acc[i][j][r];
        if constexpr (MODE == MODE_SCORES) {
          v *= scale;
          v *= prior[(long)zb * (long)M * (long)N + (long)m * N + n];
          if (mask[(long)zb * N + n] == 0) v = -1e9f;
          Cf[cfBase + (long)m * ldc + n] = v;
        } else if constexpr (MODE == MODE_F16) {
          Ch[chBase + (long)m * chRow + (long)n * chCol] = (half_t)v;
        } else if constexpr (MODE == MODE_F16_BIAS_RELU) {
          v += bias[n];
          v = v > 0.0f ? v : 0.0f;
          Ch[chBase + (long)m * chRow + (long)n * chCol] = (half_t)v;
        } else {  // MODE_F32_RESID
          if (bias) v += bias[n];
          v += resid[(long)m * ldr + n];
          Cf[cfBase + (long)m * ldc + n] = v;
        }
      }
    }
  }
}

// ---------------------------------------------------------------------------
// LayerNorm over D=1024 (one block per row), fp32 in -> f16 out
// ---------------------------------------------------------------------------
__global__ __launch_bounds__(256) void layernorm_f16_kernel(
    const float* __restrict__ x, const float* __restrict__ g,
    const float* __restrict__ bta, half_t* __restrict__ out) {
  const long row = blockIdx.x;
  const float* p = x + row * (long)Dc;
  const int tid = threadIdx.x;
  float vals[4];
  float s = 0.0f, s2 = 0.0f;
  #pragma unroll
  for (int i = 0; i < 4; ++i) {
    const float v = p[tid + i * 256];
    vals[i] = v;
    s += v;
    s2 += v * v;
  }
  __shared__ float rs[256], rq[256];
  rs[tid] = s; rq[tid] = s2;
  __syncthreads();
  for (int off = 128; off > 0; off >>= 1) {
    if (tid < off) { rs[tid] += rs[tid + off]; rq[tid] += rq[tid + off]; }
    __syncthreads();
  }
  const float mu  = rs[0] * (1.0f / Dc);
  const float var = rq[0] * (1.0f / Dc) - mu * mu;
  const float inv = rsqrtf(var + 1e-6f);
  half_t* o = out + row * (long)Dc;
  #pragma unroll
  for (int i = 0; i < 4; ++i) {
    const int idx = tid + i * 256;
    o[idx] = (half_t)((vals[i] - mu) * inv * g[idx] + bta[idx]);
  }
}

// ---------------------------------------------------------------------------
// Row softmax over 2048 columns, register-resident (1 read + 1 write of row)
// ---------------------------------------------------------------------------
__global__ __launch_bounds__(256) void softmax_kernel(float* __restrict__ attn) {
  const long row = blockIdx.x;
  float* p = attn + row * (long)Lc;
  const int tid = threadIdx.x;
  float vals[8];
  float lmax = -3.4e38f;
  #pragma unroll
  for (int i = 0; i < 8; ++i) {
    vals[i] = p[tid + i * 256];
    lmax = fmaxf(lmax, vals[i]);
  }
  __shared__ float red[256];
  red[tid] = lmax;
  __syncthreads();
  for (int off = 128; off > 0; off >>= 1) {
    if (tid < off) red[tid] = fmaxf(red[tid], red[tid + off]);
    __syncthreads();
  }
  const float m = red[0];
  __syncthreads();
  float ls = 0.0f;
  #pragma unroll
  for (int i = 0; i < 8; ++i) {
    vals[i] = __expf(vals[i] - m);
    ls += vals[i];
  }
  red[tid] = ls;
  __syncthreads();
  for (int off = 128; off > 0; off >>= 1) {
    if (tid < off) red[tid] += red[tid + off];
    __syncthreads();
  }
  const float inv = 1.0f / red[0];
  #pragma unroll
  for (int i = 0; i < 8; ++i) p[tid + i * 256] = vals[i] * inv;
}

// ---------------------------------------------------------------------------
// fp32 -> f16 elementwise (weight conversion)
// ---------------------------------------------------------------------------
__global__ __launch_bounds__(256) void f32_to_f16_kernel(
    const float* __restrict__ in, half_t* __restrict__ out, long n) {
  long i = (long)blockIdx.x * blockDim.x + threadIdx.x;
  const long stride = (long)gridDim.x * blockDim.x;
  for (; i < n; i += stride) out[i] = (half_t)in[i];
}

// ---------------------------------------------------------------------------
extern "C" void kernel_launch(void* const* d_in, const int* in_sizes, int n_in,
                              void* d_out, int out_size, void* d_ws, size_t ws_size,
                              hipStream_t stream) {
  (void)in_sizes; (void)n_in; (void)out_size; (void)ws_size;

  const float* src   = (const float*)d_in[0];
  const int*   amask = (const int*)d_in[1];
  const float* prior = (const float*)d_in[2];
  const float* ln1_g = (const float*)d_in[3];
  const float* ln1_b = (const float*)d_in[4];
  const float* wq    = (const float*)d_in[5];
  const float* wk    = (const float*)d_in[6];
  const float* wv    = (const float*)d_in[7];
  const float* fc_w  = (const float*)d_in[8];
  const float* ln2_g = (const float*)d_in[9];
  const float* ln2_b = (const float*)d_in[10];
  const float* w1_w  = (const float*)d_in[11];
  const float* w1_b  = (const float*)d_in[12];
  const float* w2_w  = (const float*)d_in[13];
  const float* w2_b  = (const float*)d_in[14];

  const long BL = (long)Bc * Lc;  // 8192

  // outputs: y [B*L*D] then attn [B*H*L*L]
  float* out_y = (float*)d_out;
  float* attn  = out_y + BL * (long)Dc;

  // workspace carve-up (halves first, then one fp32 buffer)
  half_t* w = (half_t*)d_ws;
  half_t* wq_h = w; w += (long)Dc * Dc;
  half_t* wk_h = w; w += (long)Dc * Dc;
  half_t* wv_h = w; w += (long)Dc * Dc;
  half_t* fc_h = w; w += (long)Dc * Dc;
  half_t* w1_h = w; w += (long)DFFc * Dc;
  half_t* w2_h = w; w += (long)Dc * DFFc;
  half_t* xh   = w; w += BL * (long)Dc;   // LN1 out; reused as LN2 out
  half_t* qh   = w; w += BL * (long)Dc;   // Q; reused as attn output o
  half_t* kh   = w; w += BL * (long)Dc;   // K
  half_t* vT   = w; w += BL * (long)Dc;   // V, stored [b][h*64+d][L]
  half_t* hh   = w; w += BL * (long)DFFc; // FFN hidden
  float*  yf   = (float*)w;               // post-attn residual y (fp32)

  half_t* oh = qh;  // reuse
  half_t* zh = xh;  // reuse

  // ---- 1. weight conversions ---------------------------------------------
  f32_to_f16_kernel<<<2048, 256, 0, stream>>>(wq,   wq_h, (long)Dc * Dc);
  f32_to_f16_kernel<<<2048, 256, 0, stream>>>(wk,   wk_h, (long)Dc * Dc);
  f32_to_f16_kernel<<<2048, 256, 0, stream>>>(wv,   wv_h, (long)Dc * Dc);
  f32_to_f16_kernel<<<2048, 256, 0, stream>>>(fc_w, fc_h, (long)Dc * Dc);
  f32_to_f16_kernel<<<4096, 256, 0, stream>>>(w1_w, w1_h, (long)DFFc * Dc);
  f32_to_f16_kernel<<<4096, 256, 0, stream>>>(w2_w, w2_h, (long)Dc * DFFc);

  // ---- 2. LN1 -------------------------------------------------------------
  layernorm_f16_kernel<<<(unsigned)BL, 256, 0, stream>>>(src, ln1_g, ln1_b, xh);

  // ---- 3. Q = x @ wq^T  [8192,1024] --------------------------------------
  gemm_wmma<MODE_F16, false><<<dim3(Dc / TN, BL / TM, 1), 256, 0, stream>>>(
      xh, Dc, 0, 0, wq_h, Dc, 0, 0,
      nullptr, 0, 0, 0, qh, Dc, 1, 0, 0,
      nullptr, nullptr, 0, nullptr, nullptr,
      1, 0.0f, (int)BL, Dc, Dc);
  // ---- 4. K ---------------------------------------------------------------
  gemm_wmma<MODE_F16, false><<<dim3(Dc / TN, BL / TM, 1), 256, 0, stream>>>(
      xh, Dc, 0, 0, wk_h, Dc, 0, 0,
      nullptr, 0, 0, 0, kh, Dc, 1, 0, 0,
      nullptr, nullptr, 0, nullptr, nullptr,
      1, 0.0f, (int)BL, Dc, Dc);
  // ---- 5. V (stored transposed per batch: vT[b][n][l]) --------------------
  gemm_wmma<MODE_F16, false><<<dim3(Dc / TN, Lc / TM, Bc), 256, 0, stream>>>(
      xh, Dc, (long)Lc * Dc, 0, wv_h, Dc, 0, 0,
      nullptr, 0, 0, 0, vT, /*chRow*/ 1, /*chCol*/ Lc, /*chSB*/ (long)Dc * Lc, 0,
      nullptr, nullptr, 0, nullptr, nullptr,
      1, 0.0f, Lc, Dc, Dc);

  // ---- 6. scores = (q/sqrt(dk)) @ k^T * prior, mask -> attn (fp32) --------
  gemm_wmma<MODE_SCORES, false><<<dim3(Lc / TN, Lc / TM, Bc * Hc), 256, 0, stream>>>(
      qh, Dc, (long)Lc * Dc, DKc, kh, Dc, (long)Lc * Dc, DKc,
      attn, Lc, (long)Hc * Lc * Lc, (long)Lc * Lc,
      nullptr, 0, 0, 0, 0,
      nullptr, nullptr, 0, prior, amask,
      Hc, 0.125f, Lc, Lc, DKc);

  // ---- 7. softmax over keys ----------------------------------------------
  softmax_kernel<<<(unsigned)(Bc * Hc * Lc), 256, 0, stream>>>(attn);

  // ---- 8. o = attn @ v  (attn fp32 -> f16 on the fly) ---------------------
  gemm_wmma<MODE_F16, true><<<dim3(DKc / TN, Lc / TM, Bc * Hc), 256, 0, stream>>>(
      attn, Lc, (long)Hc * Lc * Lc, (long)Lc * Lc,
      vT, Lc, (long)Dc * Lc, (long)DKc * Lc,
      nullptr, 0, 0, 0,
      oh, Dc, 1, (long)Lc * Dc, DKc,
      nullptr, nullptr, 0, nullptr, nullptr,
      Hc, 0.0f, Lc, DKc, Lc);

  // ---- 9. y = o @ fc^T + src  (fp32) --------------------------------------
  gemm_wmma<MODE_F32_RESID, false><<<dim3(Dc / TN, BL / TM, 1), 256, 0, stream>>>(
      oh, Dc, 0, 0, fc_h, Dc, 0, 0,
      yf, Dc, 0, 0, nullptr, 0, 0, 0, 0,
      nullptr, src, Dc, nullptr, nullptr,
      1, 0.0f, (int)BL, Dc, Dc);

  // ---- 10. LN2 ------------------------------------------------------------
  layernorm_f16_kernel<<<(unsigned)BL, 256, 0, stream>>>(yf, ln2_g, ln2_b, zh);

  // ---- 11. h = relu(z @ w1^T + b1) (f16) ----------------------------------
  gemm_wmma<MODE_F16_BIAS_RELU, false><<<dim3(DFFc / TN, BL / TM, 1), 256, 0, stream>>>(
      zh, Dc, 0, 0, w1_h, Dc, 0, 0,
      nullptr, 0, 0, 0, hh, DFFc, 1, 0, 0,
      w1_b, nullptr, 0, nullptr, nullptr,
      1, 0.0f, (int)BL, DFFc, Dc);

  // ---- 12. y_out = h @ w2^T + b2 + y --------------------------------------
  gemm_wmma<MODE_F32_RESID, false><<<dim3(Dc / TN, BL / TM, 1), 256, 0, stream>>>(
      hh, DFFc, 0, 0, w2_h, DFFc, 0, 0,
      out_y, Dc, 0, 0, nullptr, 0, 0, 0, 0,
      w2_b, yf, Dc, nullptr, nullptr,
      1, 0.0f, (int)BL, Dc, DFFc);
}
